// Activation2d_89593017794985
// MI455X (gfx1250) — matrix operand order repaired and checked by reading, not compiled, not run on hardware
//
#include <hip/hip_runtime.h>

// Fused alias-free Activation2d (up2x -> lrelu -> down2x) for MI455X (gfx1250).
// One wave32 computes one 16x16 output tile via 23 v_wmma_f32_16x16x32_f16.
// All stages oriented so D-stores are contiguous ds_store_b128 and operands are
// K-contiguous ds_load_b128. The polyphase band is shift-invariant (slope 8 per
// 16-tile), so ONE constant up-fragment per operand role serves every tile: the
// data gather window slides by 8*tileIdx instead. 4 constant fragments total.
// f16 data / f32 accumulate; per-wave private LDS slab, no barriers.

typedef _Float16 half_t;
typedef __attribute__((ext_vector_type(16))) _Float16 v16h;
typedef __attribute__((ext_vector_type(8)))  _Float16 v8h;
typedef __attribute__((ext_vector_type(8)))  float    v8f;
typedef __attribute__((ext_vector_type(4)))  float    v4f;

#define H_IN  256
#define W_IN  256
#define TILE  16
#define SLOPE 0.2f

// 12-tap kaiser-sinc filter (RATIO=2, KSIZE=12, cutoff=0.25, half_width=0.3),
// beta = 4.663829, normalized to sum 1 (precomputed from reference formula).
__constant__ float FILT[12] = {
   0.00202890f,  0.00938893f, -0.02554313f, -0.05765397f,
   0.12857310f,  0.44320619f,  0.44320619f,  0.12857310f,
  -0.05765397f, -0.02554313f,  0.00938893f,  0.00202890f
};

// ---- WMMA f16 fragment K-index helpers (a_k per CDNA5 ISA 7.12.2; b_k contiguous k=16*hi+h) ----
__device__ __forceinline__ int a_k(int lane, int h) {
  int v = h >> 1, hs = h & 1;
  return ((v & 3) << 1) + hs + ((v & 4) ? 16 : 0) + ((lane & 16) ? 8 : 0);
}
__device__ __forceinline__ int b_k(int lane, int h) {
  return h + ((lane & 16) ? 16 : 0);
}

__device__ __forceinline__ v8f wmma16(v16h a, v16h b, v8f c) {
  return __builtin_amdgcn_wmma_f32_16x16x32_f16(false, a, false, b,
                                                (short)0, c, false, false);
}

// Two K-contiguous 16B LDS loads -> one v16h fragment.
__device__ __forceinline__ v16h ld16(const half_t* p0, const half_t* p1) {
  v8h lo = *(const v8h*)p0;
  v8h hi = *(const v8h*)p1;
  return __builtin_shufflevector(lo, hi, 0,1,2,3,4,5,6,7,8,9,10,11,12,13,14,15);
}
// One 16B load, upper 8 elements statically zero (K>=48 tail).
__device__ __forceinline__ v16h ld8z(const half_t* p0) {
  v8h lo = *(const v8h*)p0;
  v8h z  = {};
  return __builtin_shufflevector(lo, z, 0,1,2,3,4,5,6,7,8,9,10,11,12,13,14,15);
}

// ---- Universal (tile-independent) constant band fragments.
// Up-band with data window start 8*tileIdx: tap index t = k - ((x>>1)+1+par),
// par = x&1; valid t in [0,6), else zero-padded table slot.
__device__ __forceinline__ v16h frag_upB(const half_t* ft, int lane) {
  int n = lane & 15, par = n & 1, base = (n >> 1) + 1 + par;
  v16h b;
  #pragma unroll
  for (int h = 0; h < 16; ++h) {
    unsigned t = (unsigned)(a_k(lane, h) - base);
    t = t < 6u ? t : 6u;
    b[h] = ft[2 * t + par];
  }
  return b;
}
__device__ __forceinline__ v16h frag_upA(const half_t* ft, int lane) {
  int M = lane & 15, par = M & 1, base = (M >> 1) + 1 + par;
  v16h a;
  #pragma unroll
  for (int h = 0; h < 16; ++h) {
    unsigned t = (unsigned)(a_k(lane, h) - base);
    t = t < 6u ? t : 6u;
    a[h] = ft[2 * t + par];
  }
  return a;
}
// Down-band: out o (0..15) from hi-res j (0..63): tap = f[j - 2o - 3].
__device__ __forceinline__ v16h frag_dnB(const half_t* ft, int lane, int ch) {
  int off = 32 * ch - 2 * (lane & 15) - 3;
  v16h b;
  #pragma unroll
  for (int h = 0; h < 16; ++h) {
    unsigned d = (unsigned)(b_k(lane, h) + off);
    d = d < 12u ? d : 12u;
    b[h] = ft[d];
  }
  return b;
}

// Per-wave LDS slab (halves):
//  ftab 16 @0 | ftab4 16 @16 | Xt[32c][40r] @32 (+16 zero tail @1312)
//  V[48u][40c] @1328 (+16 zero tail @3248) | Z[48u][56v] @3264
//  D1t[16n][72u] @5952 (u 48..63 zero)    | total 7104
#define SLAB 7104

__global__ __launch_bounds__(128) void
aa2d_up_lrelu_down_kernel(const float* __restrict__ x, float* __restrict__ out) {
  __shared__ half_t smem[4 * SLAB];
  const int lane = threadIdx.x & 31;
  const int wave = threadIdx.x >> 5;
  half_t* ftab  = smem + wave * SLAB;
  half_t* ftab4 = ftab + 16;            // taps * 4 (ratio^2 folded into P2)
  half_t* Xt    = ftab + 32;            // [c][r] pitch 40
  half_t* V     = ftab + 1328;          // [u][c] pitch 40
  half_t* Z     = ftab + 3264;          // [u][v] pitch 56
  half_t* D1t   = ftab + 5952;          // [n'][u] pitch 72

  const int tid = blockIdx.x * 4 + wave;   // 512 imgs * 256 tiles = 131072
  const int img = tid >> 8;
  const int ty  = (tid >> 4) & 15;
  const int tx  = tid & 15;
  const int M0  = ty * TILE, N0 = tx * TILE;
  const float* in = x + (size_t)img * (H_IN * W_IN);

  const int L  = lane & 15;
  const int hi = (lane & 16) ? 1 : 0;

  // ---- one-time init: filter tables, zero pads (sliding K-window over-read
  // must see finite data: WMMA propagates NaN even through 0.0 taps).
  {
    float fv = (lane < 12) ? FILT[lane] : 0.0f;
    if (lane < 16) { ftab[lane] = (half_t)fv; ftab4[lane] = (half_t)(4.0f * fv); }
    v8h z = {};
    if (lane < 16) *(v8h*)(D1t + lane * 72 + 48) = z;
    else           *(v8h*)(D1t + (lane - 16) * 72 + 56) = z;
    if (lane == 0) { *(v8h*)(Xt + 1280) = z; *(v8h*)(Xt + 1288) = z; }
    if (lane == 1) { *(v8h*)(V + 1920)  = z; *(v8h*)(V + 1928)  = z; }
  }

  // ---- stage X: 32x32 replicate-clamped input patch, stored transposed
  {
    int gc = N0 - 8 + lane;
    gc = gc < 0 ? 0 : (gc > W_IN - 1 ? W_IN - 1 : gc);
    const float* colp = in + gc;
    #pragma unroll
    for (int rb = 0; rb < 4; ++rb) {
      v8h pk;
      #pragma unroll
      for (int r8 = 0; r8 < 8; ++r8) {
        int gr = M0 - 8 + rb * 8 + r8;
        gr = gr < 0 ? 0 : (gr > H_IN - 1 ? H_IN - 1 : gr);
        pk[r8] = (half_t)colp[gr * W_IN];
      }
      *(v8h*)(Xt + lane * 40 + rb * 8) = pk;
    }
  }

  // ---- P1 (vertical upsample, transposed): A = Xt data (M=c, K=r, window
  // 8*it), B = universal up const. D: M=c, N=u -> store V[u][c] (b128).
  {
    v16h Bc = frag_upB(ftab, lane);
    #pragma unroll
    for (int jc = 0; jc < 2; ++jc) {
      #pragma unroll
      for (int it = 0; it < 3; ++it) {
        const half_t* rp = Xt + (16 * jc + L) * 40 + 8 * it;
        v16h a = ld16(rp + 8 * hi, rp + 16 + 8 * hi);
        v8f c = {};
        c = wmma16(a, Bc, c);
        v8h pk;
        #pragma unroll
        for (int v = 0; v < 8; ++v) pk[v] = (half_t)c[v];
        *(v8h*)(V + (16 * it + L) * 40 + 16 * jc + 8 * hi) = pk;
      }
    }
  }

  // ---- P2 (horizontal upsample * 4 + lrelu, transposed): A = universal up
  // const (taps*4), B = V data ([N=u][K=c], window 8*iz). D: M=v, N=u ->
  // lrelu(w)=max(w,0.2w), store Z[u][v] (b128).
  {
    v16h Ac = frag_upA(ftab4, lane);
    #pragma unroll
    for (int iu = 0; iu < 3; ++iu) {
      #pragma unroll
      for (int iz = 0; iz < 3; ++iz) {
        const half_t* rp = V + (16 * iu + L) * 40 + 8 * iz;
        v16h b = ld16(rp + 8 * hi, rp + 16 + 8 * hi);
        v8f c = {};
        c = wmma16(Ac, b, c);
        v8h pk;
        #pragma unroll
        for (int v = 0; v < 8; ++v) {
          float w = c[v];
          w = fmaxf(w, SLOPE * w);          // lrelu (scale already folded)
          pk[v] = (half_t)w;
        }
        *(v8h*)(Z + (16 * iu + L) * 56 + 16 * iz + 8 * hi) = pk;
      }
    }
  }

  // ---- edge fixup: downsample replicate-pad clamps hi-res col v to [0,511];
  // patch col jv maps to v = 2*N0-8+jv.
  if (N0 == 0) {
    for (int r = lane; r < 48; r += 32) {
      half_t e = Z[r * 56 + 8];
      #pragma unroll
      for (int jv = 0; jv < 8; ++jv) Z[r * 56 + jv] = e;
    }
  }
  if (N0 == W_IN - TILE) {
    for (int r = lane; r < 48; r += 32) {
      half_t e = Z[r * 56 + 39];
      #pragma unroll
      for (int jv = 40; jv < 48; ++jv) Z[r * 56 + jv] = e;
    }
  }

  // ---- shared down-band constants for P3 and P4
  v16h Bd0 = frag_dnB(ftab, lane, 0);
  v16h Bd1 = frag_dnB(ftab, lane, 1);

  // ---- P3 (horizontal downsample): A = Z data (M=u, K=v as 32 + 16+zero),
  // B = down const. D: M=u, N=n' -> store D1t[n'][u] (b128).
  {
    #pragma unroll
    for (int i = 0; i < 3; ++i) {
      const half_t* rp = Z + (16 * i + L) * 56;
      v16h a0 = ld16(rp + 16 * hi, rp + 16 * hi + 8);
      v16h a1 = hi ? (v16h){} : ld16(rp + 32, rp + 40);
      v8f c = {};
      c = wmma16(a0, Bd0, c);
      c = wmma16(a1, Bd1, c);
      v8h pk;
      #pragma unroll
      for (int v = 0; v < 8; ++v) pk[v] = (half_t)c[v];
      *(v8h*)(D1t + L * 72 + 16 * i + 8 * hi) = pk;
    }
  }

  // ---- edge fixup: vertical clamp of hi-res row u = 2*M0-8+iu to [0,511].
  if (M0 == 0 && lane < 16) {
    half_t e = D1t[lane * 72 + 8];
    #pragma unroll
    for (int u = 0; u < 8; ++u) D1t[lane * 72 + u] = e;
  }
  if (M0 == H_IN - TILE && lane < 16) {
    half_t e = D1t[lane * 72 + 39];
    #pragma unroll
    for (int u = 40; u < 48; ++u) D1t[lane * 72 + u] = e;
  }

  // ---- P4 (vertical downsample, transposed): A = D1t data (M=n', K=u,
  // padded zeros cover K>=48), B = down const. D: M=n', N=m' -> lane = output
  // row, VGPRs = 8 consecutive output cols -> two global_store_b128.
  {
    const half_t* rp = D1t + L * 72;
    v16h a0 = ld16(rp + 16 * hi,      rp + 16 * hi + 8);
    v16h a1 = ld16(rp + 32 + 16 * hi, rp + 40 + 16 * hi);
    v8f o = {};
    o = wmma16(a0, Bd0, o);
    o = wmma16(a1, Bd1, o);

    float* rowp = out + (size_t)img * (H_IN * W_IN) + (M0 + L) * W_IN + N0 + 8 * hi;
    v4f s0 = {o[0], o[1], o[2], o[3]};
    v4f s1 = {o[4], o[5], o[6], o[7]};
    *(v4f*)rowp = s0;
    *(v4f*)(rowp + 4) = s1;
  }
}

extern "C" void kernel_launch(void* const* d_in, const int* in_sizes, int n_in,
                              void* d_out, int out_size, void* d_ws, size_t ws_size,
                              hipStream_t stream) {
  (void)in_sizes; (void)n_in; (void)d_ws; (void)ws_size; (void)out_size;
  const float* x = (const float*)d_in[0];
  float* out = (float*)d_out;
  // 512 images * 256 tiles / 4 waves-per-block = 32768 blocks of 128 threads.
  hipLaunchKernelGGL(aa2d_up_lrelu_down_kernel, dim3(32768), dim3(128), 0, stream,
                     x, out);
}
